// GCN_33878702031414
// MI455X (gfx1250) — matrix operand order repaired and checked by reading, not compiled, not run on hardware
//
#include <hip/hip_runtime.h>
#include <hip/hip_bf16.h>
#include <stdint.h>

#define NUM_NODES 6150
#define NUM_EDGES 98400
#define KSEL      615
#define BATCH     256
#define KP1       1280   /* 1230 padded up to multiple of 32 */
#define N1        1024
#define K2        1024
#define N2        512
#define EPS_BN    1e-5f
#define CDIV(a,b) (((a)+(b)-1)/(b))

typedef __attribute__((ext_vector_type(16))) __bf16 v16bf;
typedef __attribute__((ext_vector_type(8)))  float  v8f;

__device__ __forceinline__ unsigned short f2bf(float f) {
    union { float f; unsigned u; } x; x.f = f;
    unsigned u = x.u;
    return (unsigned short)((u + 0x7fffu + ((u >> 16) & 1u)) >> 16);
}

/* ------------------------------------------------------------------ */
__global__ void zero_u32(unsigned* __restrict__ p, long n) {
    long i = (long)blockIdx.x * blockDim.x + threadIdx.x;
    if (i < n) p[i] = 0u;
}

/* scatter-add over edges: agg[b][dst][c] += x[b][src][c]  (L2-resident atomics) */
__global__ void scatter_add(const float* __restrict__ x,
                            const int* __restrict__ src, const int* __restrict__ dst,
                            float* __restrict__ agg) {
    long tid = (long)blockIdx.x * blockDim.x + threadIdx.x;
    if (tid >= (long)BATCH * NUM_EDGES) return;
    int e = (int)(tid % NUM_EDGES);
    int b = (int)(tid / NUM_EDGES);
    int s = src[e], d = dst[e];
    const float* xr = x   + (long)b * NUM_NODES * 2;
    float*       ar = agg + (long)b * NUM_NODES * 2;
    atomicAdd(&ar[2 * d + 0], xr[2 * s + 0]);
    atomicAdd(&ar[2 * d + 1], xr[2 * s + 1]);
}

/* y = relu(agg @ w_rel + x @ w_root + b), per (batch,node), 2 channels */
__global__ void conv1_combine(const float* __restrict__ x, const float* __restrict__ agg,
                              const float* __restrict__ wr, const float* __restrict__ wo,
                              const float* __restrict__ bia, float* __restrict__ y) {
    long t = (long)blockIdx.x * blockDim.x + threadIdx.x;
    if (t >= (long)BATCH * NUM_NODES) return;
    float a0 = agg[2*t], a1 = agg[2*t+1];
    float x0 = x[2*t],   x1 = x[2*t+1];
    float y0 = a0*wr[0] + a1*wr[2] + x0*wo[0] + x1*wo[2] + bia[0];
    float y1 = a0*wr[1] + a1*wr[3] + x0*wo[1] + x1*wo[3] + bia[1];
    y[2*t]   = fmaxf(y0, 0.f);
    y[2*t+1] = fmaxf(y1, 0.f);
}

/* score = tanh(agg @ w_rel_p + x1 @ w_root_p + b_p), 1 channel */
__global__ void score_kernel(const float* __restrict__ x1, const float* __restrict__ agg,
                             const float* __restrict__ wrp, const float* __restrict__ worp,
                             const float* __restrict__ bp, float* __restrict__ score) {
    long t = (long)blockIdx.x * blockDim.x + threadIdx.x;
    if (t >= (long)BATCH * NUM_NODES) return;
    float s = agg[2*t]*wrp[0] + agg[2*t+1]*wrp[1]
            + x1[2*t]*worp[0] + x1[2*t+1]*worp[1] + bp[0];
    score[t] = tanhf(s);
}

/* per-batch top-K by iterative masked argmax; gathers x1*val into bf16 H rows */
__global__ void __launch_bounds__(1024)
topk_gather(float* __restrict__ score, const float* __restrict__ x1,
            unsigned short* __restrict__ Hbf) {
    int b = blockIdx.x;
    float* sc = score + (long)b * NUM_NODES;
    const float* xr = x1 + (long)b * NUM_NODES * 2;
    unsigned short* hr = Hbf + (long)b * KP1;
    __shared__ unsigned long long red[1024];
    int tid = threadIdx.x;
    for (int k = 0; k < KSEL; ++k) {
        unsigned long long best = 0ull;
        for (int i = tid; i < NUM_NODES; i += 1024) {
            unsigned u = __float_as_uint(sc[i]);
            u ^= (u & 0x80000000u) ? 0xFFFFFFFFu : 0x80000000u;   /* order-preserving */
            unsigned long long key = ((unsigned long long)u << 32)
                                   | (unsigned)(NUM_NODES - 1 - i); /* tie -> smaller i */
            if (key > best) best = key;
        }
        red[tid] = best;
        __syncthreads();
        for (int s = 512; s > 0; s >>= 1) {
            if (tid < s && red[tid + s] > red[tid]) red[tid] = red[tid + s];
            __syncthreads();
        }
        if (tid == 0) {
            unsigned long long key = red[0];
            int idx = NUM_NODES - 1 - (int)(key & 0xFFFFFFFFu);
            unsigned u = (unsigned)(key >> 32);
            u ^= (u & 0x80000000u) ? 0x80000000u : 0xFFFFFFFFu;
            float val = __uint_as_float(u);
            sc[idx] = -3.4e38f;                    /* mask out */
            hr[2*k]     = f2bf(xr[2*idx]     * val);
            hr[2*k + 1] = f2bf(xr[2*idx + 1] * val);
        }
        __syncthreads();
    }
}

/* fp32 weight [N][Ksrc] -> bf16 [N][Kdst], zero-padded in K */
__global__ void conv_weight(const float* __restrict__ w, unsigned short* __restrict__ o,
                            int N, int Ksrc, int Kdst) {
    long t = (long)blockIdx.x * blockDim.x + threadIdx.x;
    if (t >= (long)N * Kdst) return;
    int k = (int)(t % Kdst);
    int n = (int)(t / Kdst);
    o[t] = (k < Ksrc) ? f2bf(w[(long)n * Ksrc + k]) : (unsigned short)0;
}

/* C[M][N] (f32) = A[M][Kp] (bf16, row-major) x B[N][Kp]^T (bf16, row-major over K).
 * One wave = one 16x16 tile; 8 waves / block. Direct global b128 loads per the
 * ISA 7.12.2 bf16 fragment layouts; accumulate with v_wmma_f32_16x16x32_bf16. */
__global__ void __launch_bounds__(256)
gemm_bf16_wmma(const unsigned short* __restrict__ A, const unsigned short* __restrict__ B,
               float* __restrict__ C, int M, int N, int Kp) {
    int wave = threadIdx.x >> 5;
    int lane = threadIdx.x & 31;
    int tile = blockIdx.x * 8 + wave;
    int tn = tile % (N >> 4);
    int tm = tile / (N >> 4);
    if (tm * 16 >= M) return;                       /* wave-uniform */
    int m0 = tm << 4, n0 = tn << 4;
    int lrow = lane & 15;
    int lhi  = lane >> 4;
    /* A fragment: row m0+lrow, K chunks [lhi*8..+7] and [16+lhi*8..+7] */
    const unsigned short* arow = A + (long)(m0 + lrow) * Kp + lhi * 8;
    /* B fragment: col n0+lrow (= weight row), K chunks [lhi*8..+7] and [16+lhi*8..+7] */
    const unsigned short* brow = B + (long)(n0 + lrow) * Kp + lhi * 8;
    v8f acc = {};
    for (int k0 = 0; k0 < Kp; k0 += 32) {
        union { uint4 q[2]; v16bf v; } a, bb;
        a.q[0]  = *(const uint4*)(arow + k0);
        a.q[1]  = *(const uint4*)(arow + k0 + 16);
        bb.q[0] = *(const uint4*)(brow + k0);
        bb.q[1] = *(const uint4*)(brow + k0 + 16);
        __builtin_prefetch(brow + k0 + 128, 0, 1);  /* global_prefetch_b8 */
        acc = __builtin_amdgcn_wmma_f32_16x16x32_bf16(
                  false, a.v, false, bb.v, (short)0, acc, false, false);
    }
    float* crow = C + (long)m0 * N + n0 + lrow;
#pragma unroll
    for (int i = 0; i < 8; ++i)
        crow[(long)(i + 8 * lhi) * N] = acc[i];
}

/* training-mode BN over batch of relu(in); one thread per column */
__global__ void bn_relu_to_bf16(const float* __restrict__ in, const float* __restrict__ g,
                                const float* __restrict__ bt, unsigned short* __restrict__ out,
                                int ncols) {
    int c = blockIdx.x * blockDim.x + threadIdx.x;
    if (c >= ncols) return;
    float sum = 0.f, ss = 0.f;
    for (int r = 0; r < BATCH; ++r) {
        float v = fmaxf(in[(long)r * ncols + c], 0.f);
        sum += v; ss += v * v;
    }
    float m   = sum * (1.f / BATCH);
    float var = ss  * (1.f / BATCH) - m * m;
    float sc  = g[c] * rsqrtf(var + EPS_BN);
    float sh  = bt[c];
    for (int r = 0; r < BATCH; ++r) {
        float v = fmaxf(in[(long)r * ncols + c], 0.f);
        out[(long)r * ncols + c] = f2bf(sc * (v - m) + sh);
    }
}

__global__ void bn_relu_to_f32(const float* __restrict__ in, const float* __restrict__ g,
                               const float* __restrict__ bt, float* __restrict__ out,
                               int ncols) {
    int c = blockIdx.x * blockDim.x + threadIdx.x;
    if (c >= ncols) return;
    float sum = 0.f, ss = 0.f;
    for (int r = 0; r < BATCH; ++r) {
        float v = fmaxf(in[(long)r * ncols + c], 0.f);
        sum += v; ss += v * v;
    }
    float m   = sum * (1.f / BATCH);
    float var = ss  * (1.f / BATCH) - m * m;
    float sc  = g[c] * rsqrtf(var + EPS_BN);
    float sh  = bt[c];
    for (int r = 0; r < BATCH; ++r) {
        float v = fmaxf(in[(long)r * ncols + c], 0.f);
        out[(long)r * ncols + c] = sc * (v - m) + sh;
    }
}

/* fc3 ([256x512] x [512]) + BN over batch; single block, thread b = batch row */
__global__ void __launch_bounds__(256)
fc3_bn(const float* __restrict__ h, const float* __restrict__ w,
       const float* __restrict__ g, const float* __restrict__ bt, float* __restrict__ out) {
    int b = threadIdx.x;
    float acc = 0.f;
    for (int k = 0; k < N2; ++k) acc += h[(long)b * N2 + k] * w[k];
    __shared__ float sv[256], s1[256], s2[256];
    sv[b] = acc; s1[b] = acc; s2[b] = acc * acc;
    __syncthreads();
    for (int s = 128; s > 0; s >>= 1) {
        if (b < s) { s1[b] += s1[b + s]; s2[b] += s2[b + s]; }
        __syncthreads();
    }
    float m = s1[0] * (1.f / BATCH);
    float var = s2[0] * (1.f / BATCH) - m * m;
    out[b] = g[0] * (sv[b] - m) * rsqrtf(var + EPS_BN) + bt[0];
}

/* ------------------------------------------------------------------ */
extern "C" void kernel_launch(void* const* d_in, const int* in_sizes, int n_in,
                              void* d_out, int out_size, void* d_ws, size_t ws_size,
                              hipStream_t stream) {
    (void)in_sizes; (void)n_in; (void)out_size; (void)ws_size;
    const float* data    = (const float*)d_in[0];
    const int*   ei      = (const int*)  d_in[1];
    const float* w_rel1  = (const float*)d_in[2];
    const float* w_root1 = (const float*)d_in[3];
    const float* b1      = (const float*)d_in[4];
    const float* w_relp  = (const float*)d_in[5];
    const float* w_rootp = (const float*)d_in[6];
    const float* b_p     = (const float*)d_in[7];
    const float* fc1_w   = (const float*)d_in[8];
    const float* bn1_g   = (const float*)d_in[9];
    const float* bn1_b   = (const float*)d_in[10];
    const float* fc2_w   = (const float*)d_in[11];
    const float* bn2_g   = (const float*)d_in[12];
    const float* bn2_b   = (const float*)d_in[13];
    const float* fc3_w   = (const float*)d_in[14];
    const float* bn3_g   = (const float*)d_in[15];
    const float* bn3_b   = (const float*)d_in[16];
    float* out = (float*)d_out;
    const int* src = ei;
    const int* dst = ei + NUM_EDGES;

    char* ws = (char*)d_ws;
    size_t off = 0;
    auto carve = [&](size_t bytes) -> char* {
        char* p = ws + off;
        off += (bytes + 255) & ~(size_t)255;
        return p;
    };
    float*          agg   = (float*)         carve((size_t)BATCH * NUM_NODES * 2 * 4);
    float*          x1b   = (float*)         carve((size_t)BATCH * NUM_NODES * 2 * 4);
    float*          score = (float*)         carve((size_t)BATCH * NUM_NODES * 4);
    unsigned short* Hbf   = (unsigned short*)carve((size_t)BATCH * KP1 * 2);
    unsigned short* W1bf  = (unsigned short*)carve((size_t)N1 * KP1 * 2);
    unsigned short* W2bf  = (unsigned short*)carve((size_t)N2 * K2 * 2);
    float*          fc1o  = (float*)         carve((size_t)BATCH * N1 * 4);
    unsigned short* h2    = (unsigned short*)carve((size_t)BATCH * K2 * 2);
    float*          fc2o  = (float*)         carve((size_t)BATCH * N2 * 4);
    float*          h3    = (float*)         carve((size_t)BATCH * N2 * 4);

    const long aggWords = (long)BATCH * NUM_NODES * 2;
    const long hWords   = ((long)BATCH * KP1 * 2) / 4;
    const long bn = (long)BATCH * NUM_NODES;
    const long be = (long)BATCH * NUM_EDGES;

    /* graph conv 1 */
    zero_u32<<<CDIV(aggWords, 256), 256, 0, stream>>>((unsigned*)agg, aggWords);
    zero_u32<<<CDIV(hWords, 256), 256, 0, stream>>>((unsigned*)Hbf, hWords);
    scatter_add<<<CDIV(be, 256), 256, 0, stream>>>(data, src, dst, agg);
    conv1_combine<<<CDIV(bn, 256), 256, 0, stream>>>(data, agg, w_rel1, w_root1, b1, x1b);
    /* score conv + tanh */
    zero_u32<<<CDIV(aggWords, 256), 256, 0, stream>>>((unsigned*)agg, aggWords);
    scatter_add<<<CDIV(be, 256), 256, 0, stream>>>(x1b, src, dst, agg);
    score_kernel<<<CDIV(bn, 256), 256, 0, stream>>>(x1b, agg, w_relp, w_rootp, b_p, score);
    /* per-batch top-K + gather into bf16 H */
    topk_gather<<<BATCH, 1024, 0, stream>>>(score, x1b, Hbf);
    /* weight conversions */
    conv_weight<<<CDIV((long)N1 * KP1, 256), 256, 0, stream>>>(fc1_w, W1bf, N1, 1230, KP1);
    conv_weight<<<CDIV((long)N2 * K2, 256), 256, 0, stream>>>(fc2_w, W2bf, N2, K2, K2);
    /* dense head: WMMA GEMMs + BN */
    gemm_bf16_wmma<<<(BATCH / 16) * (N1 / 16) / 8, 256, 0, stream>>>(Hbf, W1bf, fc1o, BATCH, N1, KP1);
    bn_relu_to_bf16<<<CDIV(N1, 256), 256, 0, stream>>>(fc1o, bn1_g, bn1_b, h2, N1);
    gemm_bf16_wmma<<<(BATCH / 16) * (N2 / 16) / 8, 256, 0, stream>>>(h2, W2bf, fc2o, BATCH, N2, K2);
    bn_relu_to_f32<<<CDIV(N2, 256), 256, 0, stream>>>(fc2o, bn2_g, bn2_b, h3, N2);
    fc3_bn<<<1, 256, 0, stream>>>(h3, fc3_w, bn3_g, bn3_b, out);
}
